// Boundary_Enchance_57861799411778
// MI455X (gfx1250) — matrix operand order repaired and checked by reading, not compiled, not run on hardware
//
#include <hip/hip_runtime.h>
#include <hip/hip_bf16.h>
#include <math.h>

typedef __attribute__((ext_vector_type(16))) _Float16 v16h;
typedef __attribute__((ext_vector_type(8)))  float    v8f;

#define B    8
#define H    512
#define W    512
#define HW   (H*W)
#define CIN_Y 5
#define CF   16     // fuse channels
#define CC   32     // concat channels
#define COUT 16

// ---------------------------------------------------------------------------
// Kernel 1: global average pool of relu(1x1 conv 5->16) without materializing
// fuse_box. Accumulates into gap[B*16] via atomics (zeroed by memset).
// ---------------------------------------------------------------------------
__global__ void gap_kernel(const float* __restrict__ y,
                           const float* __restrict__ fuse_w,
                           const float* __restrict__ fuse_b,
                           float* __restrict__ gap) {
    __shared__ float red[256 * CF];
    const int tid = threadIdx.x;
    const int b   = blockIdx.y;

    float fw[CF * CIN_Y];
    float fb[CF];
    #pragma unroll
    for (int i = 0; i < CF * CIN_Y; ++i) fw[i] = fuse_w[i];
    #pragma unroll
    for (int i = 0; i < CF; ++i) fb[i] = fuse_b[i];

    float acc[CF];
    #pragma unroll
    for (int i = 0; i < CF; ++i) acc[i] = 0.0f;

    const int stride = gridDim.x * blockDim.x;
    for (int p = blockIdx.x * blockDim.x + tid; p < HW; p += stride) {
        float yv[CIN_Y];
        #pragma unroll
        for (int c = 0; c < CIN_Y; ++c)
            yv[c] = y[(size_t)(b * CIN_Y + c) * HW + p];
        #pragma unroll
        for (int o = 0; o < CF; ++o) {
            float s = fb[o];
            #pragma unroll
            for (int c = 0; c < CIN_Y; ++c) s += fw[o * CIN_Y + c] * yv[c];
            acc[o] += fmaxf(s, 0.0f);
        }
    }
    #pragma unroll
    for (int o = 0; o < CF; ++o) red[tid * CF + o] = acc[o];
    __syncthreads();
    if (tid < CF) {
        float s = 0.0f;
        for (int t = 0; t < 256; ++t) s += red[t * CF + tid];
        atomicAdd(&gap[b * CF + tid], s);
    }
}

// ---------------------------------------------------------------------------
// Kernel 2: SE MLP: se = sigmoid(relu(gap/HW @ W1^T) @ W2^T)   [B,5]
// ---------------------------------------------------------------------------
__global__ void se_kernel(const float* __restrict__ gap,
                          const float* __restrict__ se_w1,
                          const float* __restrict__ se_w2,
                          float* __restrict__ se) {
    const int b = threadIdx.x;
    if (b >= B) return;
    float g[CF], h[CF];
    const float inv = 1.0f / (float)HW;
    #pragma unroll
    for (int i = 0; i < CF; ++i) g[i] = gap[b * CF + i] * inv;
    #pragma unroll
    for (int j = 0; j < CF; ++j) {
        float s = 0.0f;
        #pragma unroll
        for (int i = 0; i < CF; ++i) s += g[i] * se_w1[j * CF + i];
        h[j] = fmaxf(s, 0.0f);
    }
    #pragma unroll
    for (int k = 0; k < CIN_Y; ++k) {
        float s = 0.0f;
        #pragma unroll
        for (int j = 0; j < CF; ++j) s += h[j] * se_w2[k * CF + j];
        se[b * CIN_Y + k] = 1.0f / (1.0f + __expf(-s));
    }
}

// ---------------------------------------------------------------------------
// Kernel 3: fused tail. Per block: 128-pixel strip of one row, one batch.
//   * stage concat(x, fuse(y)) 3x130x32 tile in LDS as f16 (SAME zero pad)
//   * 3x3 conv 32->16 via implicit GEMM: 9 x v_wmma_f32_16x16x32_f16 per wave
//   * epilogue: relu, 1x1 16->2, softmax, boundary head from y*se, clamp,
//     1x1 1->16 store.
// ---------------------------------------------------------------------------
#define TILE_W 128
#define TCOL   (TILE_W + 2)

__global__ __launch_bounds__(256)
void fused_conv_kernel(const float* __restrict__ x,
                       const float* __restrict__ y,
                       const float* __restrict__ fuse_w,
                       const float* __restrict__ fuse_b,
                       const float* __restrict__ se,
                       const float* __restrict__ bd_w,
                       const float* __restrict__ bd_b,
                       const float* __restrict__ fc_w,
                       const float* __restrict__ fc_b,
                       const float* __restrict__ fm_w,
                       const float* __restrict__ fm_b,
                       const float* __restrict__ cv_w,
                       const float* __restrict__ cv_b,
                       float* __restrict__ out) {
    __shared__ __attribute__((aligned(32))) _Float16 tile[3][TCOL][CC];   // 24960 B
    __shared__ __attribute__((aligned(32))) _Float16 wfrag[9][32][16];    //  9216 B
    __shared__ float fccS[8][16][16];                                     //  8192 B

    const int tid  = threadIdx.x;
    const int wave = tid >> 5;
    const int lane = tid & 31;
    const int b    = blockIdx.z;
    const int gy   = blockIdx.y;
    const int gx0  = blockIdx.x * TILE_W;

    // ---- stage x channels (0..15) of the concat tile -----------------------
    for (int idx = tid; idx < 3 * TCOL * CF; idx += 256) {
        const int pos = idx >> 4;          // /16
        const int ch  = idx & 15;
        const int r   = pos / TCOL;
        const int col = pos - r * TCOL;
        const int gyy = gy + r - 1;
        const int gxx = gx0 + col - 1;
        float v = 0.0f;
        if ((unsigned)gyy < H && (unsigned)gxx < W)
            v = x[((size_t)(b * CF + ch) * H + gyy) * W + gxx];
        tile[r][col][ch] = (_Float16)v;
    }
    // ---- stage fuse channels (16..31): recompute relu(1x1 conv 5->16) -----
    for (int pos = tid; pos < 3 * TCOL; pos += 256) {
        const int r   = pos / TCOL;
        const int col = pos - r * TCOL;
        const int gyy = gy + r - 1;
        const int gxx = gx0 + col - 1;
        if ((unsigned)gyy < H && (unsigned)gxx < W) {
            float yv[CIN_Y];
            #pragma unroll
            for (int c = 0; c < CIN_Y; ++c)
                yv[c] = y[((size_t)(b * CIN_Y + c) * H + gyy) * W + gxx];
            #pragma unroll
            for (int o = 0; o < CF; ++o) {
                float s = fuse_b[o];
                #pragma unroll
                for (int c = 0; c < CIN_Y; ++c) s += fuse_w[o * CIN_Y + c] * yv[c];
                tile[r][col][CF + o] = (_Float16)fmaxf(s, 0.0f);
            }
        } else {
            #pragma unroll
            for (int o = 0; o < CF; ++o) tile[r][col][CF + o] = (_Float16)0.0f;
        }
    }
    // ---- stage B fragments (wave32 layout: lane = N%16, K-half by lane<16) -
    for (int idx = tid; idx < 9 * 32 * 16; idx += 256) {
        const int t  = idx >> 9;           // tap 0..8
        const int rm = idx & 511;
        const int ln = rm >> 4;            // lane 0..31
        const int i  = rm & 15;            // K index within half
        const int N  = ln & 15;
        const int c  = (i < 8) ? ((ln >> 4) * 8 + i) : (16 + (ln >> 4) * 8 + (i - 8));
        const int ky = t / 3, kx = t - 3 * ky;
        wfrag[t][ln][i] = (_Float16)fc_w[((N * CC + c) * 3 + ky) * 3 + kx];
    }
    __syncthreads();

    // ---- implicit GEMM: 16 pixels (M) x 16 out-ch (N), K = 9 taps x 32 ch --
    const int m     = lane & 15;           // pixel within wave tile
    const int half  = lane >> 4;           // 0: K 0-7,16-23; 1: K 8-15,24-31
    const int choff = half * 8;            // A-layout channel base
    v8f acc = {};
    #pragma unroll
    for (int t = 0; t < 9; ++t) {
        const int ky  = t / 3, kx = t - 3 * ky;
        const int col = wave * 16 + m + kx;          // 1 + (p + kx - 1)
        const _Float16* abase = &tile[ky][col][choff];
        v16h a;
        #pragma unroll
        for (int i = 0; i < 8; ++i) { a[i] = abase[i]; a[8 + i] = abase[16 + i]; }
        v16h bm = *(const v16h*)&wfrag[t][lane][0];
        acc = __builtin_amdgcn_wmma_f32_16x16x32_f16(
                  false, a, false, bm, (short)0, acc, false, false);
    }

    // ---- spill C tile: lane holds column N=lane%16, rows (half ? 8..15 : 0..7)
    #pragma unroll
    for (int v = 0; v < 8; ++v)
        fccS[wave][half * 8 + v][m] = fmaxf(acc[v] + fc_b[m], 0.0f);
    __syncthreads();

    // ---- fused epilogue: lane handles pixel m, channels [half*8, half*8+8) -
    const int gx = gx0 + wave * 16 + m;
    float f[16];
    #pragma unroll
    for (int n = 0; n < 16; ++n) f[n] = fccS[wave][m][n];

    float m0 = fm_b[0], m1 = fm_b[1];
    #pragma unroll
    for (int n = 0; n < 16; ++n) { m0 += fm_w[n] * f[n]; m1 += fm_w[16 + n] * f[n]; }
    const float mask_scale = 1.0f / (1.0f + __expf(m0 - m1));

    float z0 = bd_b[0], z1 = bd_b[1];
    #pragma unroll
    for (int c = 0; c < CIN_Y; ++c) {
        const float yc = y[((size_t)(b * CIN_Y + c) * H + gy) * W + gx] * se[b * CIN_Y + c];
        z0 += bd_w[c] * yc;
        z1 += bd_w[CIN_Y + c] * yc;
    }
    const float bnd_scale = 1.0f / (1.0f + __expf(z0 - z1));

    const float s = fminf(fmaxf(mask_scale + bnd_scale, 0.0f), 1.0f);

    #pragma unroll
    for (int j = 0; j < 8; ++j) {
        const int ch = half * 8 + j;
        out[((size_t)(b * COUT + ch) * H + gy) * W + gx] = cv_w[ch] * s + cv_b[ch];
    }
}

// ---------------------------------------------------------------------------
extern "C" void kernel_launch(void* const* d_in, const int* in_sizes, int n_in,
                              void* d_out, int out_size, void* d_ws, size_t ws_size,
                              hipStream_t stream) {
    const float* x      = (const float*)d_in[0];
    const float* y      = (const float*)d_in[1];
    const float* fuse_w = (const float*)d_in[2];
    const float* fuse_b = (const float*)d_in[3];
    const float* se_w1  = (const float*)d_in[4];
    const float* se_w2  = (const float*)d_in[5];
    const float* bd_w   = (const float*)d_in[6];
    const float* bd_b   = (const float*)d_in[7];
    const float* fc_w   = (const float*)d_in[8];
    const float* fc_b   = (const float*)d_in[9];
    const float* fm_w   = (const float*)d_in[10];
    const float* fm_b   = (const float*)d_in[11];
    const float* cv_w   = (const float*)d_in[12];
    const float* cv_b   = (const float*)d_in[13];
    float* out = (float*)d_out;

    float* gap = (float*)d_ws;            // B*16 floats
    float* se  = gap + B * CF;            // B*5 floats

    hipMemsetAsync(d_ws, 0, (B * CF + B * CIN_Y) * sizeof(float), stream);

    gap_kernel<<<dim3(32, B), 256, 0, stream>>>(y, fuse_w, fuse_b, gap);
    se_kernel<<<1, 32, 0, stream>>>(gap, se_w1, se_w2, se);
    fused_conv_kernel<<<dim3(W / TILE_W, H, B), 256, 0, stream>>>(
        x, y, fuse_w, fuse_b, se, bd_w, bd_b, fc_w, fc_b,
        fm_w, fm_b, cv_w, cv_b, out);
}
